// Decoder_NA_21225728377200
// MI455X (gfx1250) — matrix operand order repaired and checked by reading, not compiled, not run on hardware
//
#include <hip/hip_runtime.h>

// ---------------------------------------------------------------------------
// Types for CDNA5 WMMA (wave32): bf16 16x16x32 -> f32 accumulate
// ---------------------------------------------------------------------------
typedef __attribute__((ext_vector_type(16))) __bf16 bf16x16;
typedef __attribute__((ext_vector_type(8)))  float  f32x8;
typedef __attribute__((ext_vector_type(4)))  unsigned int u32x4;

#define DM    512
#define NH    8
#define NDEPTH 3
#define NV    10000
#define NB    64
#define NIMGT 144
#define NSEQ  17
#define NFF   2048

__device__ __forceinline__ f32x8 wmma_bf16(bf16x16 a, bf16x16 b, f32x8 c) {
  // (neg_a, A, neg_b, B, c_mod, C, reuse_a, reuse_b)
  return __builtin_amdgcn_wmma_f32_16x16x32_bf16(false, a, false, b, (short)0, c,
                                                 false, false);
}

__device__ __forceinline__ f32x8 zero_f32x8() {
  f32x8 v;
#pragma unroll
  for (int i = 0; i < 8; ++i) v[i] = 0.f;
  return v;
}

// ---------------------------------------------------------------------------
// Generic GEMM: C[M,N] = act(A[M,K] @ B[K,N] + bias + res), all row-major f32,
// bf16 WMMA inner product, optional batch strides (gridDim.z).
// Tile: 64x64, K-stage 64 (two 32-K WMMA steps per barrier).
// act: 0=none, 1=relu, 2=sigmoid
// ---------------------------------------------------------------------------
__global__ __launch_bounds__(128) void gemm_kernel(
    const float* __restrict__ A, const float* __restrict__ B,
    const float* __restrict__ bias, const float* __restrict__ res,
    float* __restrict__ C,
    int M, int N, int K, int lda, int ldb, int ldc,
    long sA, long sB, long sC, int act)
{
  const int bz = blockIdx.z;
  A += (long)bz * sA;
  B += (long)bz * sB;
  C += (long)bz * sC;
  const float* resp = res ? res + (long)bz * sC : nullptr;

  const int n0 = blockIdx.x * 64;
  const int m0 = blockIdx.y * 64;
  const int tid = threadIdx.x;
  const int lane = tid & 31, w = tid >> 5;
  const int hi = lane >> 4, ln = lane & 15;

  __shared__ __bf16 As[64][72];   // A tile [m][k], stride 72 (144B = 9*16B)
  __shared__ __bf16 Bs[64][72];   // B tile transposed: Bs[n][k]

  f32x8 acc[4];
#pragma unroll
  for (int i = 0; i < 4; ++i) acc[i] = zero_f32x8();

  const int lr = tid >> 1;            // 0..63 (row of A-tile / k-row of B-tile)
  const int lcb = (tid & 1) * 32;     // 0 or 32 (32 consecutive columns)

  for (int k0 = 0; k0 < K; k0 += 64) {
    // ---- A tile 64x64: fast path = branchless float4 loads ----
    {
      const int gm = m0 + lr;
      const long arow = (long)(gm < M ? gm : (M - 1)) * lda;
      if (gm < M && (k0 + lcb + 32) <= K) {
        const float4* p = (const float4*)(A + arow + k0 + lcb);
        union { __bf16 h[32]; u32x4 q[4]; } pk;
#pragma unroll
        for (int v4 = 0; v4 < 8; ++v4) {
          const float4 f = p[v4];
          pk.h[v4 * 4 + 0] = (__bf16)f.x;
          pk.h[v4 * 4 + 1] = (__bf16)f.y;
          pk.h[v4 * 4 + 2] = (__bf16)f.z;
          pk.h[v4 * 4 + 3] = (__bf16)f.w;
        }
#pragma unroll
        for (int i = 0; i < 4; ++i)
          *(u32x4*)&As[lr][lcb + i * 8] = pk.q[i];
      } else {  // edge: clamped address + select (no exec branching)
#pragma unroll
        for (int j = 0; j < 32; ++j) {
          const int gk = k0 + lcb + j;
          const float val = A[arow + (gk < K ? gk : (K - 1))];
          As[lr][lcb + j] = (__bf16)((gm < M && gk < K) ? val : 0.f);
        }
      }
    }
    // ---- B tile 64x64, coalesced along n, stored transposed ----
    {
      const int gk = k0 + lr;
      const long brow = (long)(gk < K ? gk : (K - 1)) * ldb;
      if (gk < K && (n0 + lcb + 32) <= N) {
        const float4* p = (const float4*)(B + brow + n0 + lcb);
#pragma unroll
        for (int v4 = 0; v4 < 8; ++v4) {
          const float4 f = p[v4];
          Bs[lcb + v4 * 4 + 0][lr] = (__bf16)f.x;
          Bs[lcb + v4 * 4 + 1][lr] = (__bf16)f.y;
          Bs[lcb + v4 * 4 + 2][lr] = (__bf16)f.z;
          Bs[lcb + v4 * 4 + 3][lr] = (__bf16)f.w;
        }
      } else {
#pragma unroll
        for (int j = 0; j < 32; ++j) {
          const int gn = n0 + lcb + j;
          const float val = B[brow + (gn < N ? gn : (N - 1))];
          Bs[lcb + j][lr] = (__bf16)((gk < K && gn < N) ? val : 0.f);
        }
      }
    }
    __syncthreads();
#pragma unroll
    for (int ks = 0; ks < 2; ++ks) {
      bf16x16 af;  // A fragment: element j -> k = j + hi*16
#pragma unroll
      for (int j = 0; j < 16; ++j)
        af[j] = As[w * 16 + ln][ks * 32 + j + hi * 16];
#pragma unroll
      for (int nt = 0; nt < 4; ++nt) {
        bf16x16 bfv;  // B fragment: element j -> k = j + hi*16
#pragma unroll
        for (int j = 0; j < 16; ++j)
          bfv[j] = Bs[nt * 16 + ln][ks * 32 + j + hi * 16];
        acc[nt] = wmma_bf16(af, bfv, acc[nt]);
      }
    }
    __syncthreads();
  }

#pragma unroll
  for (int nt = 0; nt < 4; ++nt) {
    const int gn = n0 + nt * 16 + ln;
    if (gn >= N) continue;
    const float bv = bias ? bias[gn] : 0.f;
#pragma unroll
    for (int r = 0; r < 8; ++r) {
      const int gm = m0 + w * 16 + hi * 8 + r;
      if (gm >= M) continue;
      float v = acc[nt][r] + bv;
      if (resp) v += resp[(long)gm * ldc + gn];
      if (act == 1) v = fmaxf(v, 0.f);
      else if (act == 2) v = 1.f / (1.f + __expf(-v));
      C[(long)gm * ldc + gn] = v;
    }
  }
}

// ---------------------------------------------------------------------------
// Fused vocab gating: out = X + softmax(X @ embB^T) @ embB   (per 16-row tile)
// Pass 1: row maxes of logits. Pass 2: recompute logits, P=exp(s-m),
// accumulate Z and O += P @ emb_chunk in WMMA accumulators; out = X + O/Z.
// X: [9216, 512] f32, embB: [10000, 512] bf16.
// ---------------------------------------------------------------------------
__global__ __launch_bounds__(128) void vocab_gate_kernel(
    float* __restrict__ out, const float* __restrict__ X,
    const __bf16* __restrict__ embB)
{
  const int tid = threadIdx.x;
  const int lane = tid & 31;
  const int w = tid >> 5;
  const int hi = lane >> 4;
  const int ln = lane & 15;
  const int row0 = blockIdx.x * 16;

  __shared__ __bf16 Tl[16][520];   // X tile bf16, padded stride (1040B = 65*16B)
  __shared__ __bf16 Pl[16][72];    // P chunk [16 rows][64 vocab cols]
  __shared__ float Mrow[16];
  __shared__ float Zrow[16];
  __shared__ float stage[4][16];

  if (tid < 16) Mrow[tid] = -1e30f;
  {  // stage X tile as bf16 (vectorized)
    const int r = tid >> 3;
    const int cb = (tid & 7) * 64;
    const float4* src = (const float4*)(X + (size_t)(row0 + r) * DM + cb);
    union { __bf16 h[64]; u32x4 q[8]; } pk;
#pragma unroll
    for (int v4 = 0; v4 < 16; ++v4) {
      const float4 f = src[v4];
      pk.h[v4 * 4 + 0] = (__bf16)f.x;
      pk.h[v4 * 4 + 1] = (__bf16)f.y;
      pk.h[v4 * 4 + 2] = (__bf16)f.z;
      pk.h[v4 * 4 + 3] = (__bf16)f.w;
    }
#pragma unroll
    for (int i = 0; i < 8; ++i) *(u32x4*)&Tl[r][cb + i * 8] = pk.q[i];
  }
  __syncthreads();

  // ---------------- Pass 1: row maxes ----------------
  for (int v0 = 0; v0 < NV; v0 += 64) {
    const int vc = v0 + w * 16 + ln;   // this lane's vocab column
    const bool cv = vc < NV;
    f32x8 S = zero_f32x8();
    for (int k0 = 0; k0 < DM; k0 += 32) {
      bf16x16 af;
#pragma unroll
      for (int j = 0; j < 16; ++j)
        af[j] = Tl[ln][k0 + j + hi * 16];
      bf16x16 bfv;  // B = emb^T: element j -> k = k0 + hi*16 + j (contiguous 32B)
      {
        const int vcc = cv ? vc : (NV - 1);
        const u32x4* p = (const u32x4*)(embB + (size_t)vcc * DM + k0 + hi * 16);
        union { bf16x16 v; u32x4 q[2]; } u;
        u.q[0] = p[0]; u.q[1] = p[1];
        bfv = u.v;   // invalid lanes masked out at the max step below
      }
      S = wmma_bf16(af, bfv, S);
    }
#pragma unroll
    for (int r = 0; r < 8; ++r) {  // max over 16 lanes of each half
      float m = cv ? S[r] : -1e30f;
      m = fmaxf(m, __shfl_xor(m, 1));
      m = fmaxf(m, __shfl_xor(m, 2));
      m = fmaxf(m, __shfl_xor(m, 4));
      m = fmaxf(m, __shfl_xor(m, 8));
      if (ln == 0) stage[w][hi * 8 + r] = m;
    }
    __syncthreads();
    if (tid < 16) {
      float m = Mrow[tid];
#pragma unroll
      for (int ww = 0; ww < 4; ++ww) m = fmaxf(m, stage[ww][tid]);
      Mrow[tid] = m;
    }
    __syncthreads();
  }

  // ---------------- Pass 2: P, Z and O += P @ emb_chunk ----------------
  f32x8 O[8];
#pragma unroll
  for (int i = 0; i < 8; ++i) O[i] = zero_f32x8();
  float zacc[8];
#pragma unroll
  for (int r = 0; r < 8; ++r) zacc[r] = 0.f;

  for (int v0 = 0; v0 < NV; v0 += 64) {
    const int vc = v0 + w * 16 + ln;
    const bool cv = vc < NV;
    f32x8 S = zero_f32x8();
    for (int k0 = 0; k0 < DM; k0 += 32) {
      bf16x16 af;
#pragma unroll
      for (int j = 0; j < 16; ++j)
        af[j] = Tl[ln][k0 + j + hi * 16];
      bf16x16 bfv;
      {
        const int vcc = cv ? vc : (NV - 1);
        const u32x4* p = (const u32x4*)(embB + (size_t)vcc * DM + k0 + hi * 16);
        union { bf16x16 v; u32x4 q[2]; } u;
        u.q[0] = p[0]; u.q[1] = p[1];
        bfv = u.v;
      }
      S = wmma_bf16(af, bfv, S);
    }
#pragma unroll
    for (int r = 0; r < 8; ++r) {
      const float p = cv ? __expf(S[r] - Mrow[hi * 8 + r]) : 0.f;
      zacc[r] += p;
      Pl[hi * 8 + r][w * 16 + ln] = (__bf16)p;
    }
    __syncthreads();
    // O[16,512] += P[16,64] @ embChunk[64,512]; wave handles 128 D-columns
#pragma unroll
    for (int ks = 0; ks < 2; ++ks) {
      bf16x16 af;
#pragma unroll
      for (int j = 0; j < 16; ++j)
        af[j] = Pl[ln][ks * 32 + j + hi * 16];
#pragma unroll
      for (int nt = 0; nt < 8; ++nt) {
        const int col = w * 128 + nt * 16 + ln;
        bf16x16 bfv;
#pragma unroll
        for (int j = 0; j < 16; ++j) {  // branchless: clamp + select
          const int vrow = v0 + ks * 32 + hi * 16 + j;
          const __bf16 val =
              embB[(size_t)(vrow < NV ? vrow : (NV - 1)) * DM + col];
          bfv[j] = (vrow < NV) ? val : (__bf16)0.f;
        }
        O[nt] = wmma_bf16(af, bfv, O[nt]);
      }
    }
    __syncthreads();
  }

  // Z reduction across lanes and waves
#pragma unroll
  for (int r = 0; r < 8; ++r) {
    float z = zacc[r];
    z += __shfl_xor(z, 1);
    z += __shfl_xor(z, 2);
    z += __shfl_xor(z, 4);
    z += __shfl_xor(z, 8);
    if (ln == 0) stage[w][hi * 8 + r] = z;
  }
  __syncthreads();
  if (tid < 16)
    Zrow[tid] = stage[0][tid] + stage[1][tid] + stage[2][tid] + stage[3][tid];
  __syncthreads();

#pragma unroll
  for (int nt = 0; nt < 8; ++nt) {
    const int col = w * 128 + nt * 16 + ln;
#pragma unroll
    for (int r = 0; r < 8; ++r) {
      const int row = hi * 8 + r;
      const size_t idx = (size_t)(row0 + row) * DM + col;
      out[idx] = X[idx] + O[nt][r] / Zrow[row];
    }
  }
}

// ---------------------------------------------------------------------------
// Fused attention core: per (q, h, b) block, scores->softmax->weighted V sum.
// Q:[B,Lq,512], K/V:[B,Lk,512], head h = cols [h*64, h*64+64).
// ---------------------------------------------------------------------------
__global__ __launch_bounds__(128) void attn_kernel(
    const float* __restrict__ Q, const float* __restrict__ K,
    const float* __restrict__ V, float* __restrict__ O, int Lq, int Lk)
{
  const int q = blockIdx.x, h = blockIdx.y, b = blockIdx.z;
  const int tid = threadIdx.x;
  __shared__ float qs[64];
  __shared__ float sc[160];
  __shared__ float inv_z;
  const float* qp = Q + (size_t)(b * Lq + q) * DM + h * 64;
  if (tid < 64) qs[tid] = qp[tid] * 0.125f;  // 64^-0.5
  __syncthreads();
  for (int k = tid; k < Lk; k += 128) {
    const float* kp = K + (size_t)(b * Lk + k) * DM + h * 64;
    float s = 0.f;
#pragma unroll
    for (int d = 0; d < 64; ++d) s += qs[d] * kp[d];
    sc[k] = s;
  }
  __syncthreads();
  if (tid == 0) {
    float m = -1e30f;
    for (int k = 0; k < Lk; ++k) m = fmaxf(m, sc[k]);
    float z = 0.f;
    for (int k = 0; k < Lk; ++k) { const float e = __expf(sc[k] - m); sc[k] = e; z += e; }
    inv_z = 1.f / z;
  }
  __syncthreads();
  if (tid < 64) {
    float a = 0.f;
    for (int k = 0; k < Lk; ++k)
      a += sc[k] * V[(size_t)(b * Lk + k) * DM + h * 64 + tid];
    O[(size_t)(b * Lq + q) * DM + h * 64 + tid] = a * inv_z;
  }
}

// ---------------------------------------------------------------------------
// LayerNorm over last dim (512): out = LN(in + res) * g + b
// ---------------------------------------------------------------------------
__global__ __launch_bounds__(128) void ln_kernel(
    float* __restrict__ out, const float* __restrict__ in,
    const float* __restrict__ res, const float* __restrict__ g,
    const float* __restrict__ b)
{
  const int row = blockIdx.x;
  const int tid = threadIdx.x;
  __shared__ float red[128];
  const float* pi = in + (size_t)row * DM;
  const float* pr = res ? res + (size_t)row * DM : nullptr;
  float v[4];
  float s = 0.f;
#pragma unroll
  for (int i = 0; i < 4; ++i) {
    const int d = tid + i * 128;
    v[i] = pi[d] + (pr ? pr[d] : 0.f);
    s += v[i];
  }
  red[tid] = s;
  __syncthreads();
  for (int st = 64; st > 0; st >>= 1) {
    if (tid < st) red[tid] += red[tid + st];
    __syncthreads();
  }
  const float mean = red[0] * (1.f / DM);
  __syncthreads();
  s = 0.f;
#pragma unroll
  for (int i = 0; i < 4; ++i) { const float d = v[i] - mean; s += d * d; }
  red[tid] = s;
  __syncthreads();
  for (int st = 64; st > 0; st >>= 1) {
    if (tid < st) red[tid] += red[tid + st];
    __syncthreads();
  }
  const float rstd = rsqrtf(red[0] * (1.f / DM) + 1e-5f);
  float* po = out + (size_t)row * DM;
#pragma unroll
  for (int i = 0; i < 4; ++i) {
    const int d = tid + i * 128;
    po[d] = (v[i] - mean) * rstd * g[d] + b[d];
  }
}

// out = x*g + xs*(1-g)
__global__ __launch_bounds__(256) void blend_kernel(
    float* __restrict__ out, const float* __restrict__ x,
    const float* __restrict__ xs, const float* __restrict__ g, int n)
{
  const int i = blockIdx.x * 256 + threadIdx.x;
  if (i < n) {
    const float gv = g[i];
    out[i] = x[i] * gv + xs[i] * (1.f - gv);
  }
}

__global__ __launch_bounds__(256) void cvt_bf16_kernel(
    __bf16* __restrict__ out, const float* __restrict__ in, int n)
{
  const int i = blockIdx.x * 256 + threadIdx.x;
  if (i < n) out[i] = (__bf16)in[i];
}

// ---------------------------------------------------------------------------
// Host orchestration
// ---------------------------------------------------------------------------
static inline void gemm(hipStream_t s, const float* A, const float* B,
                        const float* bias, const float* res, float* C,
                        int M, int N, int K, int lda, int ldb, int ldc,
                        long sA = 0, long sB = 0, long sC = 0, int batch = 1,
                        int act = 0)
{
  dim3 grid((N + 63) / 64, (M + 63) / 64, batch);
  gemm_kernel<<<grid, 128, 0, s>>>(A, B, bias, res, C, M, N, K, lda, ldb, ldc,
                                   sA, sB, sC, act);
}

extern "C" void kernel_launch(void* const* d_in, const int* in_sizes, int n_in,
                              void* d_out, int out_size, void* d_ws, size_t ws_size,
                              hipStream_t stream)
{
  (void)in_sizes; (void)n_in; (void)out_size;

  const float* encoder_out = (const float*)d_in[0];   // [64,144,512]
  const float* wmat        = (const float*)d_in[1];   // [17,144]
  const float* i2t_w       = (const float*)d_in[2];
  const float* i2t_b       = (const float*)d_in[3];
  const float* i2t_ln_g    = (const float*)d_in[4];
  const float* i2t_ln_b    = (const float*)d_in[5];
  const float* emb         = (const float*)d_in[6];   // [10000,512]
  const float* gen_w       = (const float*)d_in[7];   // [512,10000]
  const float* gen_b       = (const float*)d_in[8];
  const float* a_w         = (const float*)d_in[9];   // [1024,512]
  const float* a_b         = (const float*)d_in[10];
  const float* fln_g       = (const float*)d_in[11];
  const float* fln_b       = (const float*)d_in[12];
  const float* sa_wq = (const float*)d_in[13]; const float* sa_bq = (const float*)d_in[14];
  const float* sa_wk = (const float*)d_in[15]; const float* sa_bk = (const float*)d_in[16];
  const float* sa_wv = (const float*)d_in[17]; const float* sa_bv = (const float*)d_in[18];
  const float* sa_wo = (const float*)d_in[19]; const float* sa_bo = (const float*)d_in[20];
  const float* ca_wq = (const float*)d_in[21]; const float* ca_bq = (const float*)d_in[22];
  const float* ca_wk = (const float*)d_in[23]; const float* ca_bk = (const float*)d_in[24];
  const float* ca_wv = (const float*)d_in[25]; const float* ca_bv = (const float*)d_in[26];
  const float* ca_wo = (const float*)d_in[27]; const float* ca_bo = (const float*)d_in[28];
  const float* ln1_g = (const float*)d_in[29]; const float* ln1_b = (const float*)d_in[30];
  const float* ln2_g = (const float*)d_in[31]; const float* ln2_b = (const float*)d_in[32];
  const float* ln3_g = (const float*)d_in[33]; const float* ln3_b = (const float*)d_in[34];
  const float* fc1_w = (const float*)d_in[35]; const float* fc1_b = (const float*)d_in[36];
  const float* fc2_w = (const float*)d_in[37]; const float* fc2_b = (const float*)d_in[38];

  const size_t nBI = (size_t)NB * NIMGT;   // 9216
  const size_t nBS = (size_t)NB * NSEQ;    // 1088

  // ---- workspace layout (256-byte aligned slabs), total ~128 MB ----
  char* ws = (char*)d_ws;
  size_t off = 0;
  auto take = [&](size_t bytes) -> void* {
    off = (off + 255) & ~(size_t)255;
    void* p = ws + off;
    off += bytes;
    return p;
  };
  __bf16* embB   = (__bf16*)take((size_t)NV * DM * 2);
  float* bufPre  = (float*)take(nBI * DM * 4);
  float* bufT    = (float*)take(nBI * DM * 4);
  float* bufW    = (float*)take(nBI * DM * 4);
  float* bufKc   = (float*)take(nBI * DM * 4);
  float* bufVc   = (float*)take(nBI * DM * 4);
  float* bufQ    = (float*)take(nBS * DM * 4);
  float* bufAttn = (float*)take(nBS * DM * 4);
  float* bufO    = (float*)take(nBS * DM * 4);
  float* bufX    = (float*)take(nBS * DM * 4);
  float* bufY    = (float*)take(nBS * DM * 4);
  float* bufSelf = (float*)take(nBS * DM * 4);
  float* bufH    = (float*)take(nBS * NFF * 4);
  float* bufG    = (float*)take(nBS * DM * 4);
  float* bufG2   = (float*)take(nBS * DM * 4);
  float* bufBl   = (float*)take(nBS * DM * 4);
  if (off > ws_size) return;  // workspace too small; bail deterministically

  // 0) bf16 copy of embedding table (kept L2-resident, halves traffic)
  {
    const int n = NV * DM;
    cvt_bf16_kernel<<<(n + 255) / 256, 256, 0, stream>>>(embB, emb, n);
  }
  // 1) text_emb = LN(encoder_out @ i2t_w + i2t_b)
  gemm(stream, encoder_out, i2t_w, i2t_b, nullptr, bufPre,
       (int)nBI, DM, DM, DM, DM, DM);
  ln_kernel<<<(int)nBI, 128, 0, stream>>>(bufT, bufPre, nullptr, i2t_ln_g, i2t_ln_b);
  // 2) word_emb = text_emb + softmax(text_emb @ emb^T) @ emb   (fused)
  vocab_gate_kernel<<<(int)(nBI / 16), 128, 0, stream>>>(bufW, bufT, embB);
  // 3) x = einsum('sn,bnd->bsd', w, word_emb): batched GEMM, batch = B
  gemm(stream, wmat, bufW, nullptr, nullptr, bufX,
       NSEQ, DM, NIMGT, NIMGT, DM, DM,
       0, (long)NIMGT * DM, (long)NSEQ * DM, NB);

  float* cur = bufX;
  float* alt = bufY;
  const size_t WSTEP = (size_t)DM * DM;
  const dim3 agrid(NSEQ, NH, NB);

  for (int i = 0; i < NDEPTH; ++i) {
    // --- self attention ---
    gemm(stream, cur, sa_wq + i * WSTEP, sa_bq + i * DM, nullptr, bufQ,  (int)nBS, DM, DM, DM, DM, DM);
    gemm(stream, cur, sa_wk + i * WSTEP, sa_bk + i * DM, nullptr, bufKc, (int)nBS, DM, DM, DM, DM, DM);
    gemm(stream, cur, sa_wv + i * WSTEP, sa_bv + i * DM, nullptr, bufVc, (int)nBS, DM, DM, DM, DM, DM);
    attn_kernel<<<agrid, 128, 0, stream>>>(bufQ, bufKc, bufVc, bufAttn, NSEQ, NSEQ);
    gemm(stream, bufAttn, sa_wo + i * WSTEP, sa_bo + i * DM, nullptr, bufSelf, (int)nBS, DM, DM, DM, DM, DM);
    ln_kernel<<<(int)nBS, 128, 0, stream>>>(alt, bufSelf, cur, ln1_g + i * DM, ln1_b + i * DM);
    { float* t = cur; cur = alt; alt = t; }
    // --- cross attention ---
    gemm(stream, cur, ca_wq + i * WSTEP, ca_bq + i * DM, nullptr, bufQ, (int)nBS, DM, DM, DM, DM, DM);
    gemm(stream, encoder_out, ca_wk + i * WSTEP, ca_bk + i * DM, nullptr, bufKc, (int)nBI, DM, DM, DM, DM, DM);
    gemm(stream, encoder_out, ca_wv + i * WSTEP, ca_bv + i * DM, nullptr, bufVc, (int)nBI, DM, DM, DM, DM, DM);
    attn_kernel<<<agrid, 128, 0, stream>>>(bufQ, bufKc, bufVc, bufAttn, NSEQ, NIMGT);
    gemm(stream, bufAttn, ca_wo + i * WSTEP, ca_bo + i * DM, nullptr, bufO, (int)nBS, DM, DM, DM, DM, DM);
    ln_kernel<<<(int)nBS, 128, 0, stream>>>(alt, bufO, cur, ln2_g + i * DM, ln2_b + i * DM);
    { float* t = cur; cur = alt; alt = t; }
    // --- FFN ---
    gemm(stream, cur, fc1_w + (size_t)i * DM * NFF, fc1_b + (size_t)i * NFF, nullptr, bufH,
         (int)nBS, NFF, DM, DM, NFF, NFF, 0, 0, 0, 1, /*relu*/1);
    gemm(stream, bufH, fc2_w + (size_t)i * NFF * DM, fc2_b + i * DM, nullptr, bufO,
         (int)nBS, DM, NFF, NFF, DM, DM);
    ln_kernel<<<(int)nBS, 128, 0, stream>>>(alt, bufO, cur, ln3_g + i * DM, ln3_b + i * DM);
    { float* t = cur; cur = alt; alt = t; }
  }

  // --- sigmoid gate over concat([x, x_self]) ---
  gemm(stream, cur, a_w, a_b, nullptr, bufG, (int)nBS, DM, DM, DM, DM, DM);
  gemm(stream, bufSelf, a_w + (size_t)DM * DM, nullptr, bufG, bufG2,
       (int)nBS, DM, DM, DM, DM, DM, 0, 0, 0, 1, /*sigmoid*/2);
  {
    const int n = (int)(nBS * DM);
    blend_kernel<<<(n + 255) / 256, 256, 0, stream>>>(bufBl, cur, bufSelf, bufG2, n);
  }
  ln_kernel<<<(int)nBS, 128, 0, stream>>>(alt, bufBl, nullptr, fln_g, fln_b);
  cur = alt;
  // --- generator -> d_out [1088, 10000] ---
  gemm(stream, cur, gen_w, gen_b, nullptr, (float*)d_out,
       (int)nBS, NV, DM, DM, NV, NV);
}
